// GatedResidualCondConv_72662256714387
// MI455X (gfx1250) — matrix operand order repaired and checked by reading, not compiled, not run on hardware
//
#include <hip/hip_runtime.h>
#include <hip/hip_bf16.h>

typedef __attribute__((ext_vector_type(16))) __bf16 v16bf;
typedef __attribute__((ext_vector_type(8)))  __bf16 v8bf;
typedef __attribute__((ext_vector_type(8)))  float  v8f;

#define B_   8
#define R_   128
#define C_   128
#define D_   128
#define S_   256
#define T_   16384
#define DIL_ 64
#define TP_  16320          // T - DIL
#define TN_  64             // time tile per workgroup (255 tiles)

// packed-weight workspace layout (offsets in bf16 elements; 512 elems per fragment)
#define FRAG_ELEMS 512
#define NFRAG_FILT (8*12)   // 8 M-tiles x 12 K-steps (K=384)
#define NFRAG_GATE (8*12)
#define NFRAG_RES  (8*4)    // 8 M-tiles x 4 K-steps (K=128)
#define NFRAG_SKP  (16*4)   // 16 M-tiles x 4 K-steps
#define OFF_FILT 0
#define OFF_GATE (OFF_FILT + NFRAG_FILT*FRAG_ELEMS)
#define OFF_RES  (OFF_GATE + NFRAG_GATE*FRAG_ELEMS)
#define OFF_SKP  (OFF_RES  + NFRAG_RES *FRAG_ELEMS)
#define WS_ELEMS (OFF_SKP  + NFRAG_SKP *FRAG_ELEMS)   // 147456 bf16 = 288 KB

__device__ __forceinline__ unsigned pack_bf16x2(float a, float b) {
    union { __bf16 h[2]; unsigned u; } p;
    p.h[0] = (__bf16)a;
    p.h[1] = (__bf16)b;
    return p.u;
}

// ---------------------------------------------------------------------------
// Pack fp32 weights into bf16 WMMA A-fragment order.
// A-fragment (16x32 bf16, wave32): lane l -> M = l&15 (+16*mtile), g = l>>4,
// element e -> K = kstep*32 + g*16 + e.
// ---------------------------------------------------------------------------
__global__ __launch_bounds__(256) void pack_weights(
    const float* __restrict__ w_sig,  const float* __restrict__ w_gate,
    const float* __restrict__ w_psig, const float* __restrict__ w_pgate,
    const float* __restrict__ w_res,  const float* __restrict__ w_skp,
    __bf16* __restrict__ ws)
{
    int idx = blockIdx.x * 256 + threadIdx.x;
    if (idx >= WS_ELEMS) return;

    int region, local;
    if      (idx < OFF_GATE) { region = 0; local = idx - OFF_FILT; }
    else if (idx < OFF_RES)  { region = 1; local = idx - OFF_GATE; }
    else if (idx < OFF_SKP)  { region = 2; local = idx - OFF_RES;  }
    else                     { region = 3; local = idx - OFF_SKP;  }

    int f    = local >> 9;          // fragment index within region
    int rem  = local & 511;
    int lane = rem >> 4;
    int e    = rem & 15;
    int hn   = lane & 15;
    int g    = lane >> 4;

    float v;
    if (region < 2) {
        // filt/gate: K = 384 = [w2[:,:,0] | w2[:,:,1] | w_proj]
        int m = f / 12, k = f % 12;
        int row = m * 16 + hn;                         // D row
        int K   = k * 32 + g * 16 + e;
        const float* w2 = (region == 0) ? w_sig  : w_gate;   // (D,R,2)
        const float* wp = (region == 0) ? w_psig : w_pgate;  // (D,C)
        if      (K < 128) v = w2[row * 256 + K * 2];
        else if (K < 256) v = w2[row * 256 + (K - 128) * 2 + 1];
        else              v = wp[row * 128 + (K - 256)];
    } else {
        int m = f >> 2, k = f & 3;
        int row = m * 16 + hn;                         // R or S row
        int K   = k * 32 + g * 16 + e;
        v = (region == 2) ? w_res[row * 128 + K]       // (R,D)
                          : w_skp[row * 128 + K];      // (S,D)
    }
    ws[idx] = (__bf16)v;
}

// ---------------------------------------------------------------------------
// Main fused kernel: one workgroup = (batch b, 64-wide time tile).
// 8 waves; wave w owns M-tile rows [16w,16w+16) of D in stage 1,
// rows [16w,16w+16) of R and [32w,32w+32) of S in stage 2.
// ---------------------------------------------------------------------------
__global__ __launch_bounds__(256) void wavenet_layer(
    const float* __restrict__ x,     const float* __restrict__ cond,
    const float* __restrict__ b_sig, const float* __restrict__ b_gate,
    const __bf16* __restrict__ ws,
    float* __restrict__ out_sig, float* __restrict__ out_skp)
{
    // transposed bf16 staging: in_t[t][k], k = [x0 rows | x1 rows | cond rows]
    __shared__ __align__(32) __bf16 in_t[TN_][384];   // 48 KB
    __shared__ __align__(32) __bf16 z_t [TN_][128];   // 16 KB

    const int tid  = threadIdx.x;
    const int wave = tid >> 5;
    const int lane = tid & 31;
    const int g    = lane >> 4;
    const int hn   = lane & 15;
    const int b    = blockIdx.y;
    const int t0   = blockIdx.x * TN_;
    const int mbase = wave * 16;

    // ---- stage input tiles (fp32 -> packed bf16x2, transposed into LDS) ----
    const float* x0p = x    + (size_t)b * R_ * T_ + t0;
    const float* x1p = x0p  + DIL_;
    const float* cp  = cond + (size_t)b * C_ * T_ + t0 + DIL_;
    for (int i = tid; i < 64 * TN_; i += 256) {     // 64 row-pairs x 64 cols
        int rp = i >> 6;
        int tl = i & (TN_ - 1);
        int r  = rp * 2;
        size_t o0 = (size_t)r * T_ + tl;
        size_t o1 = o0 + T_;
        *(unsigned*)&in_t[tl][r]       = pack_bf16x2(x0p[o0], x0p[o1]);
        *(unsigned*)&in_t[tl][128 + r] = pack_bf16x2(x1p[o0], x1p[o1]);
        *(unsigned*)&in_t[tl][256 + r] = pack_bf16x2(cp[o0],  cp[o1]);
    }
    __syncthreads();

    const v16bf* wsv = (const v16bf*)ws;   // 16 bf16 per slot; frag = 32 slots

    // ---- stage 1: pre-activations, K = 384, bf16 WMMA, f32 accumulate ----
    v8f accF[4] = {};
    v8f accG[4] = {};
    {
        const v16bf* pF = wsv + (OFF_FILT / FRAG_ELEMS + wave * 12) * 32 + lane;
        const v16bf* pG = wsv + (OFF_GATE / FRAG_ELEMS + wave * 12) * 32 + lane;
        for (int k = 0; k < 12; ++k) {
            v16bf aF = *pF;  pF += 32;
            v16bf aG = *pG;  pG += 32;
#pragma unroll
            for (int nt = 0; nt < 4; ++nt) {
                v16bf bfr = *(const v16bf*)&in_t[nt * 16 + hn][k * 32 + g * 16];
                accF[nt] = __builtin_amdgcn_wmma_f32_16x16x32_bf16(
                    false, aF, false, bfr, (short)0, accF[nt], false, false);
                accG[nt] = __builtin_amdgcn_wmma_f32_16x16x32_bf16(
                    false, aG, false, bfr, (short)0, accG[nt], false, false);
            }
        }
    }

    // ---- activation: z = tanh(filt+b) * sigmoid(gate+b); pack bf16 -> LDS ----
    float bs[8], bg[8];
    {
        const float4* bsv4 = (const float4*)(b_sig  + mbase + g * 8);
        const float4* bgv4 = (const float4*)(b_gate + mbase + g * 8);
        *(float4*)&bs[0] = bsv4[0];  *(float4*)&bs[4] = bsv4[1];
        *(float4*)&bg[0] = bgv4[0];  *(float4*)&bg[4] = bgv4[1];
    }
    const float L2E = 1.4426950408889634f;   // log2(e)
#pragma unroll
    for (int nt = 0; nt < 4; ++nt) {
        v8bf zv;
#pragma unroll
        for (int i = 0; i < 8; ++i) {
            float fv = accF[nt][i] + bs[i];
            float gv = accG[nt][i] + bg[i];
            // tanh(f) = 1 - 2*e/(1+e), e = 2^(-2*log2(e)*f), f clamped
            float fc = __builtin_fminf(__builtin_fmaxf(fv, -30.f), 30.f);
            float ef = __builtin_amdgcn_exp2f(-2.f * L2E * fc);
            float th = 1.f - 2.f * ef * __builtin_amdgcn_rcpf(1.f + ef);
            // sigmoid(g) = 1/(1 + 2^(-log2(e)*g)), g clamped
            float gc = __builtin_fminf(__builtin_fmaxf(gv, -60.f), 60.f);
            float eg = __builtin_amdgcn_exp2f(-L2E * gc);
            float sg = __builtin_amdgcn_rcpf(1.f + eg);
            zv[i] = (__bf16)(th * sg);
        }
        // D-fragment: lane holds 8 contiguous d rows at fixed t -> one 16B DS store
        *(v8bf*)&z_t[nt * 16 + hn][mbase + g * 8] = zv;
    }
    __syncthreads();

    // ---- stage 2: sig = w_res*z (+x1), skp = w_skp*z; K = 128 ----
    v8f accS[4]  = {};
    v8f accK0[4] = {};
    v8f accK1[4] = {};
    {
        const v16bf* pR  = wsv + (OFF_RES / FRAG_ELEMS + wave * 4) * 32 + lane;
        const v16bf* pK0 = wsv + (OFF_SKP / FRAG_ELEMS + (2 * wave)     * 4) * 32 + lane;
        const v16bf* pK1 = wsv + (OFF_SKP / FRAG_ELEMS + (2 * wave + 1) * 4) * 32 + lane;
        for (int k = 0; k < 4; ++k) {
            v16bf aR  = *pR;   pR  += 32;
            v16bf aK0 = *pK0;  pK0 += 32;
            v16bf aK1 = *pK1;  pK1 += 32;
#pragma unroll
            for (int nt = 0; nt < 4; ++nt) {
                v16bf bz = *(const v16bf*)&z_t[nt * 16 + hn][k * 32 + g * 16];
                accS[nt]  = __builtin_amdgcn_wmma_f32_16x16x32_bf16(
                    false, aR,  false, bz, (short)0, accS[nt],  false, false);
                accK0[nt] = __builtin_amdgcn_wmma_f32_16x16x32_bf16(
                    false, aK0, false, bz, (short)0, accK0[nt], false, false);
                accK1[nt] = __builtin_amdgcn_wmma_f32_16x16x32_bf16(
                    false, aK1, false, bz, (short)0, accK1[nt], false, false);
            }
        }
    }

    // ---- outputs (fp32): base pointer per lane + constant immediate offsets ----
    float*       sigp  = out_sig + (size_t)(b * R_ + mbase + g * 8) * TP_ + t0 + hn;
    const float* xresp = x       + (size_t)(b * R_ + mbase + g * 8) * T_  + t0 + DIL_ + hn;
    float*       skpp  = out_skp + (size_t)(b * S_ + wave * 32 + g * 8) * TP_ + t0 + hn;
#pragma unroll
    for (int i = 0; i < 8; ++i) {
#pragma unroll
        for (int nt = 0; nt < 4; ++nt) {
            sigp[(size_t)i * TP_ + nt * 16] =
                accS[nt][i] + xresp[(size_t)i * T_ + nt * 16];
            skpp[(size_t)i        * TP_ + nt * 16] = accK0[nt][i];
            skpp[(size_t)(i + 16) * TP_ + nt * 16] = accK1[nt][i];
        }
    }
}

// ---------------------------------------------------------------------------
extern "C" void kernel_launch(void* const* d_in, const int* in_sizes, int n_in,
                              void* d_out, int out_size, void* d_ws, size_t ws_size,
                              hipStream_t stream)
{
    const float* x       = (const float*)d_in[0];
    const float* cond    = (const float*)d_in[1];
    const float* w_sig   = (const float*)d_in[2];
    const float* b_sig   = (const float*)d_in[3];
    const float* w_gate  = (const float*)d_in[4];
    const float* b_gate  = (const float*)d_in[5];
    const float* w_psig  = (const float*)d_in[6];
    const float* w_pgate = (const float*)d_in[7];
    const float* w_res   = (const float*)d_in[8];
    const float* w_skp   = (const float*)d_in[9];

    float* out_sig = (float*)d_out;
    float* out_skp = out_sig + (size_t)B_ * R_ * TP_;
    __bf16* ws = (__bf16*)d_ws;   // needs 288 KB

    pack_weights<<<(WS_ELEMS + 255) / 256, 256, 0, stream>>>(
        w_sig, w_gate, w_psig, w_pgate, w_res, w_skp, ws);

    dim3 grid(TP_ / TN_, B_);     // 255 x 8 workgroups
    wavenet_layer<<<grid, 256, 0, stream>>>(
        x, cond, b_sig, b_gate, ws, out_sig, out_skp);
}
